// MultiHeadAttention_80161269612662
// MI455X (gfx1250) — matrix operand order repaired and checked by reading, not compiled, not run on hardware
//
#include <hip/hip_runtime.h>
#include <hip/hip_bf16.h>

// ---------------- CDNA5 WMMA plumbing ----------------
typedef __attribute__((ext_vector_type(16))) __bf16 v16bf;
typedef __attribute__((ext_vector_type(8)))  float  v8f;

union Frag {
  v16bf v;
  unsigned short u[16];
  uint4 q[2];
};

static __device__ __forceinline__ unsigned short f2bf(float x) {
  unsigned int u = __float_as_uint(x);
  unsigned int r = ((u >> 16) & 1u) + 0x7FFFu;   // round-to-nearest-even
  return (unsigned short)((u + r) >> 16);
}
static __device__ __forceinline__ float bf2f(unsigned short h) {
  return __uint_as_float(((unsigned int)h) << 16);
}
static __device__ __forceinline__ v8f zero8() {
  v8f r;
#pragma unroll
  for (int i = 0; i < 8; ++i) r[i] = 0.0f;
  return r;
}
static __device__ __forceinline__ v8f wmma_bf16(const Frag& a, const Frag& b, v8f c) {
  // v_wmma_f32_16x16x32_bf16: D = A(16x32) * B(32x16) + C(16x16 f32)
  return __builtin_amdgcn_wmma_f32_16x16x32_bf16(false, a.v, false, b.v,
                                                 (short)0, c, false, false);
}

#define LDIM 1024
#define NB   4
#define EDIM 1024
#define HNUM 16
#define HD   64
#define E3   3072

// ---------------- elementwise f32 -> bf16 convert ----------------
__global__ void k_convert_bf16(const float* __restrict__ src,
                               unsigned short* __restrict__ dst, long n) {
  long idx = (long)blockIdx.x * blockDim.x + threadIdx.x;
  if (idx < n) dst[idx] = f2bf(src[idx]);
}

// ---------------- weight transpose + bf16 convert ----------------
// src: (J,K) f32 row-major  ->  dst: (K,J) bf16 row-major
__global__ void k_transpose_bf16(const float* __restrict__ src,
                                 unsigned short* __restrict__ dst,
                                 int J, int K) {
  long idx = (long)blockIdx.x * blockDim.x + threadIdx.x;
  long total = (long)J * K;
  if (idx >= total) return;
  int j = (int)(idx / K);
  int k = (int)(idx % K);
  dst[(long)k * J + j] = f2bf(src[idx]);
}

// ---------------- QKV projection ----------------
// C[m,c] = sum_e query[m,e] * W_in[c,e] + b_in[c];  m = n*1024+l  (M=4096, K=1024, N=3072)
// Scatter into Qb[b,h,s,d] (scaled 0.125), Kt[b,h,d,s], Vb[b,h,s,d] as bf16,
// where b=m>>10, s=m&1023, h=c/192.
// block = 256 thr (8 waves, 4x2), block tile 128x128, wave tile 32x64.
__global__ void k_qkv_proj(const unsigned short* __restrict__ Qf,   // (4096,1024) bf16
                           const unsigned short* __restrict__ WtIn, // (1024,3072) bf16
                           const float* __restrict__ b_in,
                           unsigned short* __restrict__ Qb,
                           unsigned short* __restrict__ Kt,
                           unsigned short* __restrict__ Vb) {
  const int wave = threadIdx.x >> 5, lane = threadIdx.x & 31;
  const int wm = wave & 3, wn = wave >> 2;
  const int m0 = blockIdx.x * 128 + wm * 32;
  const int n0 = blockIdx.y * 128 + wn * 64;
  const int mloc = lane & 15;
  const int khalf = (lane >> 4) << 3;
  const int mrow_off = (lane >> 4) << 3;
  const int nloc = lane & 15;

  v8f acc[2][4];
#pragma unroll
  for (int i = 0; i < 2; ++i)
#pragma unroll
    for (int j = 0; j < 4; ++j) acc[i][j] = zero8();

  for (int k = 0; k < EDIM; k += 32) {
    Frag a[2], b[4];
#pragma unroll
    for (int i = 0; i < 2; ++i) {
      const unsigned short* ap = Qf + (size_t)(m0 + i * 16 + mloc) * EDIM + k + khalf;
      a[i].q[0] = *(const uint4*)(ap);
      a[i].q[1] = *(const uint4*)(ap + 16);
    }
#pragma unroll
    for (int j = 0; j < 4; ++j) {
      const unsigned short* bp = WtIn + (size_t)k * E3 + n0 + j * 16 + nloc;
#pragma unroll
      for (int s = 0; s < 16; ++s)
        b[j].u[s] = bp[(size_t)(khalf + (s & 7) + ((s >> 3) << 4)) * E3];
    }
    // prefetch next k-step (gfx1250 global_prefetch_b8; speculative, OOB-safe)
    __builtin_prefetch((const void*)(WtIn + (size_t)(k + 32 + lane) * E3 + n0), 0, 0);
    __builtin_prefetch((const void*)(Qf + (size_t)(m0 + mloc) * EDIM + k + 32), 0, 0);
#pragma unroll
    for (int i = 0; i < 2; ++i)
#pragma unroll
      for (int j = 0; j < 4; ++j)
        acc[i][j] = wmma_bf16(a[i], b[j], acc[i][j]);
  }

#pragma unroll
  for (int i = 0; i < 2; ++i)
#pragma unroll
    for (int j = 0; j < 4; ++j)
#pragma unroll
      for (int r = 0; r < 8; ++r) {
        int m = m0 + i * 16 + r + mrow_off;
        int c = n0 + j * 16 + nloc;
        float val = acc[i][j][r] + b_in[c];
        int bb = m >> 10, s = m & 1023;
        int h = c / 192, rr = c - h * 192;
        int bh = bb * HNUM + h;
        if (rr < 64) {
          Qb[((size_t)bh * LDIM + s) * HD + rr] = f2bf(val * 0.125f);
        } else if (rr < 128) {
          Kt[((size_t)bh * HD + (rr - 64)) * LDIM + s] = f2bf(val);
        } else {
          Vb[((size_t)bh * LDIM + s) * HD + (rr - 128)] = f2bf(val);
        }
      }
}

// ---------------- fused attention ----------------
// grid 256 = (b in 4) x (64 row-chunks of 16); block 256 thr (8 waves).
// dyn LDS: S f32[16][1024] | P bf16[16][1024] | Mean f32[16][1024] |
//          red f32[256] | invs f32[16] | Pt f32[8][16][16]
__global__ void k_attention(const unsigned short* __restrict__ Qb,
                            const unsigned short* __restrict__ Kt,
                            const unsigned short* __restrict__ Vb,
                            unsigned short* __restrict__ Ctx,  // (4,1024,1024) bf16
                            float* __restrict__ out2) {        // (4,1024,1024) f32
  extern __shared__ char smem[];
  float* S = (float*)smem;                                        // 64 KB
  unsigned short* P = (unsigned short*)(smem + 16 * 1024 * 4);    // 32 KB
  float* Mean = (float*)(smem + 16 * 1024 * 4 + 16 * 1024 * 2);   // 64 KB
  float* red = Mean + 16 * 1024;                                  // 1 KB
  float* invs = red + 256;                                        // 64 B
  float* Pt = invs + 16;                                          // 8 KB

  const int b = blockIdx.x >> 6;
  const int row0 = (blockIdx.x & 63) * 16;
  const int tid = threadIdx.x, wave = tid >> 5, lane = tid & 31;
  const int mloc = lane & 15, khalf = (lane >> 4) << 3;
  const int mrow_off = (lane >> 4) << 3, nloc = lane & 15;
  const int row = tid >> 4, l16 = tid & 15;

  for (int h = 0; h < HNUM; ++h) {
    const int bh = b * HNUM + h;
    const unsigned short* Qbh = Qb + (size_t)bh * LDIM * HD;
    const unsigned short* Ktb = Kt + (size_t)bh * HD * LDIM;
    const unsigned short* Vbh = Vb + (size_t)bh * LDIM * HD;

    // ---- scores S = (q*scale) @ k^T : 16 rows x 1024 cols, K = 64 ----
    for (int nt = 0; nt < 8; ++nt) {
      int n0 = wave * 128 + nt * 16;
      v8f acc = zero8();
#pragma unroll
      for (int ks = 0; ks < HD; ks += 32) {
        Frag a, bf;
        const unsigned short* ap = Qbh + (size_t)(row0 + mloc) * HD + ks + khalf;
        a.q[0] = *(const uint4*)(ap);
        a.q[1] = *(const uint4*)(ap + 16);
        const unsigned short* bp = Ktb + (size_t)ks * LDIM + n0 + nloc;
#pragma unroll
        for (int s = 0; s < 16; ++s)
          bf.u[s] = bp[(size_t)(khalf + (s & 7) + ((s >> 3) << 4)) * LDIM];
        acc = wmma_bf16(a, bf, acc);
      }
#pragma unroll
      for (int r = 0; r < 8; ++r)
        S[(r + mrow_off) * 1024 + n0 + nloc] = acc[r];
    }
    __syncthreads();

    // ---- softmax over each of the 16 rows (16 threads per row) ----
    float mx = -3.0e38f;
    for (int j = l16; j < LDIM; j += 16) mx = fmaxf(mx, S[row * 1024 + j]);
    red[row * 16 + l16] = mx;
    __syncthreads();
    if (l16 == 0) {
      float m2 = red[row * 16];
      for (int i = 1; i < 16; ++i) m2 = fmaxf(m2, red[row * 16 + i]);
      red[row * 16] = m2;
    }
    __syncthreads();
    mx = red[row * 16];
    __syncthreads();
    float sum = 0.0f;
    for (int j = l16; j < LDIM; j += 16) {
      float e = __expf(S[row * 1024 + j] - mx);
      P[row * 1024 + j] = f2bf(e);
      sum += e;
    }
    red[row * 16 + l16] = sum;
    __syncthreads();
    if (l16 == 0) {
      float s2 = 0.0f;
      for (int i = 0; i < 16; ++i) s2 += red[row * 16 + i];
      invs[row] = 1.0f / s2;
    }
    __syncthreads();
    const float iv = invs[row];

    // ---- accumulate head-mean of probabilities ----
    if (h == 0) {
      for (int j = l16; j < LDIM; j += 16)
        Mean[row * 1024 + j] = bf2f(P[row * 1024 + j]) * iv;
    } else {
      for (int j = l16; j < LDIM; j += 16)
        Mean[row * 1024 + j] += bf2f(P[row * 1024 + j]) * iv;
    }

    // ---- ctx = P @ V : 16 x 64, K = 1024; all 8 waves (K split in halves) ----
    {
      int n0 = (wave & 3) * 16;
      int kb = (wave >> 2) * 512;
      v8f acc = zero8();
      for (int kk = 0; kk < 512; kk += 32) {
        int k = kb + kk;
        Frag a, bf;
        const unsigned short* ap = P + (size_t)mloc * 1024 + k + khalf;
        a.q[0] = *(const uint4*)(ap);
        a.q[1] = *(const uint4*)(ap + 16);
        const unsigned short* bp = Vbh + (size_t)k * HD + n0 + nloc;
#pragma unroll
        for (int s = 0; s < 16; ++s)
          bf.u[s] = bp[(size_t)(khalf + (s & 7) + ((s >> 3) << 4)) * HD];
        acc = wmma_bf16(a, bf, acc);
      }
#pragma unroll
      for (int r = 0; r < 8; ++r)
        Pt[wave * 256 + (r + mrow_off) * 16 + nloc] = acc[r];
    }
    __syncthreads();

    // ---- combine K-halves, normalize, write ctx ----
    for (int idx = tid; idx < 1024; idx += 256) {
      int m = idx >> 6, n = idx & 63;
      int w = n >> 4, nl = n & 15;
      float val = (Pt[w * 256 + m * 16 + nl] + Pt[(w + 4) * 256 + m * 16 + nl]) * invs[m];
      Ctx[((size_t)b * LDIM + row0 + m) * EDIM + h * HD + n] = f2bf(val);
    }
    __syncthreads();  // S/P/Pt reused next head
  }

  // ---- write attn mean: out2[b, row0+row, :] ----
  for (int j = l16; j < LDIM; j += 16)
    out2[((size_t)b * LDIM + row0 + row) * LDIM + j] = Mean[row * 1024 + j] * 0.0625f;
}

// ---------------- output projection ----------------
// out1[(l*4+b), c] = sum_e Ctx[(b,l),e] * W_out[c,e] + b_out[c]; M=4096,K=1024,N=1024
// block tile 128x128, wave tile 32x64
__global__ void k_out_proj(const unsigned short* __restrict__ Ctx,
                           const unsigned short* __restrict__ WtOut, // (1024,1024) bf16
                           const float* __restrict__ b_out,
                           float* __restrict__ out1) {
  const int wave = threadIdx.x >> 5, lane = threadIdx.x & 31;
  const int wm = wave & 3, wn = wave >> 2;
  const int m0 = blockIdx.x * 128 + wm * 32;
  const int n0 = blockIdx.y * 128 + wn * 64;
  const int mloc = lane & 15, khalf = (lane >> 4) << 3;
  const int mrow_off = (lane >> 4) << 3, nloc = lane & 15;

  v8f acc[2][4];
#pragma unroll
  for (int i = 0; i < 2; ++i)
#pragma unroll
    for (int j = 0; j < 4; ++j) acc[i][j] = zero8();

  for (int k = 0; k < EDIM; k += 32) {
    Frag a[2], b[4];
#pragma unroll
    for (int i = 0; i < 2; ++i) {
      const unsigned short* ap = Ctx + (size_t)(m0 + i * 16 + mloc) * EDIM + k + khalf;
      a[i].q[0] = *(const uint4*)(ap);
      a[i].q[1] = *(const uint4*)(ap + 16);
    }
#pragma unroll
    for (int j = 0; j < 4; ++j) {
      const unsigned short* bp = WtOut + (size_t)k * EDIM + n0 + j * 16 + nloc;
#pragma unroll
      for (int s = 0; s < 16; ++s)
        b[j].u[s] = bp[(size_t)(khalf + (s & 7) + ((s >> 3) << 4)) * EDIM];
    }
    __builtin_prefetch((const void*)(WtOut + (size_t)(k + 32 + lane) * EDIM + n0), 0, 0);
    __builtin_prefetch((const void*)(Ctx + (size_t)(m0 + mloc) * EDIM + k + 32), 0, 0);
#pragma unroll
    for (int i = 0; i < 2; ++i)
#pragma unroll
      for (int j = 0; j < 4; ++j)
        acc[i][j] = wmma_bf16(a[i], b[j], acc[i][j]);
  }

#pragma unroll
  for (int i = 0; i < 2; ++i)
#pragma unroll
    for (int j = 0; j < 4; ++j)
#pragma unroll
      for (int r = 0; r < 8; ++r) {
        int m = m0 + i * 16 + r + mrow_off;
        int c = n0 + j * 16 + nloc;
        int bb = m >> 10, l = m & 1023;
        out1[((size_t)l * NB + bb) * EDIM + c] = acc[i][j][r] + b_out[c];
      }
}

// ---------------- launcher ----------------
extern "C" void kernel_launch(void* const* d_in, const int* in_sizes, int n_in,
                              void* d_out, int out_size, void* d_ws, size_t ws_size,
                              hipStream_t stream) {
  const float* query = (const float*)d_in[0];
  const float* W_in  = (const float*)d_in[3];
  const float* b_in  = (const float*)d_in[4];
  const float* W_out = (const float*)d_in[5];
  const float* b_out = (const float*)d_in[6];

  unsigned short* Qb    = (unsigned short*)d_ws;                 // 4M bf16
  unsigned short* Kt    = Qb   + (size_t)4 * 1024 * 1024;        // 4M bf16
  unsigned short* Vb    = Kt   + (size_t)4 * 1024 * 1024;        // 4M bf16
  unsigned short* Ctx   = Vb   + (size_t)4 * 1024 * 1024;        // 4M bf16
  unsigned short* WtIn  = Ctx  + (size_t)4 * 1024 * 1024;        // 3M bf16
  unsigned short* WtOut = WtIn + (size_t)3 * 1024 * 1024;        // 1M bf16
  unsigned short* Qf    = WtOut + (size_t)1 * 1024 * 1024;       // 4M bf16

  float* out1 = (float*)d_out;                                   // (L,N,E)
  float* out2 = out1 + (size_t)4 * 1024 * 1024;                  // (N,L,S)

  k_convert_bf16<<<(4 * 1024 * 1024 + 255) / 256, 256, 0, stream>>>(
      query, Qf, (long)4 * 1024 * 1024);
  k_transpose_bf16<<<(3 * 1024 * 1024 + 255) / 256, 256, 0, stream>>>(W_in, WtIn, E3, EDIM);
  k_transpose_bf16<<<(1024 * 1024 + 255) / 256, 256, 0, stream>>>(W_out, WtOut, EDIM, EDIM);

  dim3 g1(32, 24);  // M/128, 3072/128
  k_qkv_proj<<<g1, 256, 0, stream>>>(Qf, WtIn, b_in, Qb, Kt, Vb);

  size_t smem = (size_t)16 * 1024 * 4 + 16 * 1024 * 2 + 16 * 1024 * 4 +
                256 * 4 + 16 * 4 + 8 * 256 * 4;
  k_attention<<<dim3(256), dim3(256), smem, stream>>>(Qb, Kt, Vb, Ctx, out2);

  dim3 g2(32, 8);   // M/128, 1024/128
  k_out_proj<<<g2, 256, 0, stream>>>(Ctx, WtOut, b_out, out1);
}
